// DceLoss_14869176778889
// MI455X (gfx1250) — matrix operand order repaired and checked by reading, not compiled, not run on hardware
//
#include <hip/hip_runtime.h>
#include <hip/hip_bf16.h>

// ---- CDNA5 WMMA types (probe-verified) -------------------------------------
typedef __attribute__((ext_vector_type(16))) _Float16 v16h;
typedef __attribute__((ext_vector_type(8)))  float    v8f;

static __device__ __forceinline__ v8f wmma_f16(v16h a, v16h b, v8f c) {
  // (neg_a, A, neg_b, B, c_mod, C, reuse_a, reuse_b)
  return __builtin_amdgcn_wmma_f32_16x16x32_f16(false, a, false, b, (short)0, c,
                                                false, false);
}

#ifndef DCE_D
#define DCE_D 128   // feature dim (compile-time)
#endif
#ifndef DCE_C
#define DCE_C 1024  // num prototypes (compile-time)
#endif

#define LOG2E_F 1.4426950408889634f
#define LN2_F   0.6931471805599453f

// ---- Kernel 1: prototypes f32 -> f16, and y_sq[c] = ||proto_c||^2 ----------
__global__ __launch_bounds__(DCE_D) void dce_prep(const float* __restrict__ protos,
                                                  _Float16* __restrict__ protoH,
                                                  float* __restrict__ ysq) {
  const int c = blockIdx.x;
  const int t = threadIdx.x;                       // 128 threads, one per dim
  float v = protos[(size_t)c * DCE_D + t];
  protoH[(size_t)c * DCE_D + t] = (_Float16)v;
  __shared__ float sd[DCE_D];
  sd[t] = v * v;
  __syncthreads();
  for (int s = DCE_D / 2; s > 0; s >>= 1) {
    if (t < s) sd[t] += sd[t + s];
    __syncthreads();
  }
  if (t == 0) ysq[c] = sd[0];
}

// ---- Kernel 2: fused GEMM + sqrt + streaming softmax + NLL per 32-row wave --
__global__ __launch_bounds__(256) void dce_main(const float* __restrict__ feats,
                                                const _Float16* __restrict__ protoH,
                                                const float* __restrict__ ysq,
                                                const int* __restrict__ labels,
                                                float* __restrict__ partial) {
  __shared__ float s_ysq[DCE_C];
  const int tid = threadIdx.x;
  for (int i = tid; i < DCE_C; i += 256) s_ysq[i] = ysq[i];
  __syncthreads();

  const int wave = tid >> 5;
  const int lane = tid & 31;
  const int hf   = lane >> 4;    // lane half selects K sub-range in fragments
  const int r16  = lane & 15;
  const int rowBase = blockIdx.x * 256 + wave * 32;

  // --- Preload A fragments (2 M-tiles x 4 K-chunks) + per-row ||x||^2 -------
  // A fragments here use contiguous K (same order as B): lanes 0-15 row M=lane hold K = k0+0..15;
  // lanes 16-31 row M=lane-16 hold K = k0+16..31.
  v16h  A[2][4];
  float xsq[2];
  int   lab[2];
#pragma unroll
  for (int t = 0; t < 2; ++t) {
    const int row = rowBase + t * 16 + r16;
    const float* fr = feats + (size_t)row * DCE_D;
    lab[t] = labels[row];
    float acc = 0.f;
#pragma unroll
    for (int k = 0; k < 4; ++k) {
      const int base = 32 * k + hf * 16;
      float4 f0 = *(const float4*)(fr + base);
      float4 f1 = *(const float4*)(fr + base + 4);
      float4 g0 = *(const float4*)(fr + base + 8);
      float4 g1 = *(const float4*)(fr + base + 12);
      v16h a;
      a[0]=(_Float16)f0.x; a[1]=(_Float16)f0.y; a[2]=(_Float16)f0.z; a[3]=(_Float16)f0.w;
      a[4]=(_Float16)f1.x; a[5]=(_Float16)f1.y; a[6]=(_Float16)f1.z; a[7]=(_Float16)f1.w;
      a[8]=(_Float16)g0.x; a[9]=(_Float16)g0.y; a[10]=(_Float16)g0.z; a[11]=(_Float16)g0.w;
      a[12]=(_Float16)g1.x; a[13]=(_Float16)g1.y; a[14]=(_Float16)g1.z; a[15]=(_Float16)g1.w;
      A[t][k] = a;
      acc += f0.x*f0.x + f0.y*f0.y + f0.z*f0.z + f0.w*f0.w;
      acc += f1.x*f1.x + f1.y*f1.y + f1.z*f1.z + f1.w*f1.w;
      acc += g0.x*g0.x + g0.y*g0.y + g0.z*g0.z + g0.w*g0.w;
      acc += g1.x*g1.x + g1.y*g1.y + g1.z*g1.z + g1.w*g1.w;
    }
    xsq[t] = acc + __shfl_xor(acc, 16, 32);  // lanes L and L+16 cover disjoint K
  }

  // Broadcast x_sq / label to the C/D-fragment layout: element v in lane L is
  // row m = v + 8*(L>>4); row m's value lives in lane m (and m+16).
  float xsqE[2][8];
  int   labE[2][8];
#pragma unroll
  for (int t = 0; t < 2; ++t)
#pragma unroll
    for (int v = 0; v < 8; ++v) {
      const int src = v + hf * 8;
      xsqE[t][v] = __shfl(xsq[t], src, 32);
      labE[t][v] = __shfl(lab[t], src, 32);
    }

  float se[2][8];  // per-(row,lane) partial sum of exp(logit)
  float tg[2][8];  // label logit (exactly one lane/tile hits per row)
#pragma unroll
  for (int t = 0; t < 2; ++t)
#pragma unroll
    for (int v = 0; v < 8; ++v) { se[t][v] = 0.f; tg[t][v] = 0.f; }

  // --- Sweep all C prototype columns in 16-wide tiles, 1-deep SW pipeline ---
  // B fragments here use contiguous K (same order as A): lane = column n; lanes 0-15 hold K=k0+0..15,
  // lanes 16-31 hold K=k0+16..31, packed 2 per VGPR.
  const int NT = DCE_C / 16;                      // 64 tiles
  v16h  bcur[4];
  float yqcur;
  {
    const _Float16* pb = protoH + (size_t)r16 * DCE_D + hf * 16;
#pragma unroll
    for (int k = 0; k < 4; ++k) bcur[k] = *(const v16h*)(pb + k * 32);
    yqcur = s_ysq[r16];
  }

  for (int ct = 0; ct < NT; ++ct) {
    // Prefetch next tile's B fragments + y_sq (wraps on last iter; harmless).
    const int ctn  = (ct + 1 < NT) ? (ct + 1) : 0;
    const int coln = ctn * 16 + r16;
    v16h  bnext[4];
    {
      const _Float16* pb = protoH + (size_t)coln * DCE_D + hf * 16;
#pragma unroll
      for (int k = 0; k < 4; ++k) bnext[k] = *(const v16h*)(pb + k * 32);
    }
    const float yqnext = s_ysq[coln];

    const int col = ct * 16 + r16;                // this lane's class column
    v8f acc0 = {};
    v8f acc1 = {};
#pragma unroll
    for (int k = 0; k < 4; ++k) {
      acc0 = wmma_f16(A[0][k], bcur[k], acc0);    // reuse B across both M-tiles
      acc1 = wmma_f16(A[1][k], bcur[k], acc1);
    }

#pragma unroll
    for (int t = 0; t < 2; ++t) {
      const v8f acc = t ? acc1 : acc0;
#pragma unroll
      for (int v = 0; v < 8; ++v) {
        float xy = acc[v];
        float sq = fmaxf(xsqE[t][v] + yqcur - 2.0f * xy, 0.0f);
        float s  = __builtin_amdgcn_sqrtf(sq);            // raw v_sqrt_f32
        // exp(-s) = exp2(-s*log2(e)); args in [-45,0] -> no guard needed
        se[t][v] += __builtin_amdgcn_exp2f(-s * LOG2E_F); // raw v_exp_f32
        if (col == labE[t][v]) tg[t][v] = -s;             // hit exactly once
      }
    }

#pragma unroll
    for (int k = 0; k < 4; ++k) bcur[k] = bnext[k];
    yqcur = yqnext;
  }

  // --- Reduce across the 16 lanes of each half-wave -------------------------
#pragma unroll
  for (int t = 0; t < 2; ++t)
#pragma unroll
    for (int v = 0; v < 8; ++v) {
      float s = se[t][v], g = tg[t][v];
#pragma unroll
      for (int m = 1; m <= 8; m <<= 1) {
        s += __shfl_xor(s, m, 32);
        g += __shfl_xor(g, m, 32);
      }
      se[t][v] = s; tg[t][v] = g;
    }

  // lane 0 holds rows {v}, lane 16 holds rows {v+8}, for both M-tiles.
  float nll = 0.f;
  if (r16 == 0) {
#pragma unroll
    for (int t = 0; t < 2; ++t)
#pragma unroll
      for (int v = 0; v < 8; ++v)
        // ln(sumexp) - logit_label ; ln via raw v_log_f32 (log2) * ln2
        nll += LN2_F * __builtin_amdgcn_logf(se[t][v]) - tg[t][v];
  }
  nll += __shfl_xor(nll, 16, 32);
  if (lane == 0) partial[blockIdx.x * 8 + wave] = nll;
}

// ---- Kernel 3: deterministic fixed-order reduction + mean ------------------
__global__ __launch_bounds__(256) void dce_reduce(const float* __restrict__ partial,
                                                  float* __restrict__ out,
                                                  int n, float invN) {
  __shared__ float sd[256];
  const int t = threadIdx.x;
  float s = 0.f;
  for (int i = t; i < n; i += 256) s += partial[i];
  sd[t] = s;
  __syncthreads();
  for (int k = 128; k > 0; k >>= 1) {
    if (t < k) sd[t] += sd[t + k];
    __syncthreads();
  }
  if (t == 0) out[0] = sd[0] * invN;
}

extern "C" void kernel_launch(void* const* d_in, const int* in_sizes, int n_in,
                              void* d_out, int out_size, void* d_ws, size_t ws_size,
                              hipStream_t stream) {
  const float* feats  = (const float*)d_in[0];
  const float* protos = (const float*)d_in[1];
  const int*   labels = (const int*)d_in[2];
  float*       out    = (float*)d_out;

  const int N = in_sizes[0] / DCE_D;             // 262144
  // workspace layout: [protoH f16 C*D][ysq f32 C][partials f32 N/32]
  char* ws = (char*)d_ws;
  _Float16* protoH  = (_Float16*)ws;
  float*    ysq     = (float*)(ws + (size_t)DCE_C * DCE_D * sizeof(_Float16));
  float*    partial = (float*)(ws + (size_t)DCE_C * DCE_D * sizeof(_Float16)
                                  + (size_t)DCE_C * sizeof(float));

  dce_prep<<<DCE_C, DCE_D, 0, stream>>>(protos, protoH, ysq);

  const int blocks = N / 256;                    // 256 rows per block (8 waves)
  dce_main<<<blocks, 256, 0, stream>>>(feats, protoH, ysq, labels, partial);

  dce_reduce<<<1, 256, 0, stream>>>(partial, out, blocks * 8, 1.0f / (float)N);
}
